// ReRanker_11630771438108
// MI455X (gfx1250) — matrix operand order repaired and checked by reading, not compiled, not run on hardware
//
#include <hip/hip_runtime.h>
#include <hip/hip_bf16.h>

#define DEV __device__ __forceinline__

// ---------------- problem dimensions ----------------
#define B_   32
#define TC_  400
#define TQ_  64
#define KK_  10
#define L_   96
#define H_   150
#define D_   300
#define EMB_ 300
#define G3_  450      // 3*H
#define N_   320      // B*K
#define FD_  1200     // 4*D
// padded dims for WMMA (K mult of 32, weight-rows mult of 16)
#define EMBP 320
#define G3P  464
#define DP   304
// register blocking: each wave computes (16*WM_) x (16*WN_) output
#define WM_  2
#define WN_  4

typedef __attribute__((ext_vector_type(16))) __bf16 v16bf;
typedef __attribute__((ext_vector_type(8)))  float  v8f;
typedef __attribute__((ext_vector_type(8)))  unsigned short us8;

union FragBf { us8 h[2]; v16bf v; };

DEV unsigned short f2bf(float x) {
  unsigned int u = __float_as_uint(x);
  u += 0x7FFFu + ((u >> 16) & 1u);          // round-to-nearest-even
  return (unsigned short)(u >> 16);
}
DEV float sigm(float x) { return 1.f / (1.f + __expf(-x)); }

// ---------------- weight pad+convert: f32 (R,C) -> bf16 (Rp,Cp) zero-padded --------
__global__ void k_convert_pad(const float* __restrict__ src, unsigned short* __restrict__ dst,
                              int R, int C, int Rp, int Cp) {
  long long i = (long long)blockIdx.x * blockDim.x + threadIdx.x;
  long long total = (long long)Rp * Cp;
  if (i >= total) return;
  int r = (int)(i / Cp), c = (int)(i % Cp);
  float v = (r < R && c < C) ? src[(long long)r * C + c] : 0.f;
  dst[i] = f2bf(v);
}

// ---------------- embedding gather: tokens (Bb,T) -> X bf16 (T*Bb, EMBP), mask (Bb,T) ----
__global__ void k_gather_embed(const int* __restrict__ tokens, const float* __restrict__ emb,
                               unsigned short* __restrict__ X, float* __restrict__ mask,
                               int T, int Bb) {
  long long i = (long long)blockIdx.x * blockDim.x + threadIdx.x;
  long long total = (long long)T * Bb * EMBP;
  if (i >= total) return;
  int c = (int)(i % EMBP);
  long long row = i / EMBP;
  int b = (int)(row % Bb), t = (int)(row / Bb);
  int tok = tokens[b * T + t];
  float v = (c < EMB_) ? emb[(long long)tok * EMB_ + c] : 0.f;
  X[row * EMBP + c] = f2bf(v);
  if (c == 0) mask[b * T + t] = (tok != 0) ? 1.f : 0.f;
}

// ---------------- generic batched WMMA GEMM, 32x64 per wave ----------------
// C[m,n] = sum_k A[m,k]*W[n,k] (+bias[n]); bf16 in, f32 out.
// Kp mult of 32. A rows = M (mult of 32). W has Wrows padded rows (mult of 16);
// edge tiles clamp the W-row index so fragment loads never go OOB; stores guarded.
// One wave per 32x64 block; 8 waves per block; batch = blockIdx.y (W ptr uses batch/bDiv).
__global__ void k_gemm_bf16(const unsigned short* __restrict__ A,
                            const unsigned short* __restrict__ W,
                            float* __restrict__ C,
                            const float* __restrict__ bias,
                            int M, int N, int Wrows, int Kp, int lda, int ldb, int ldc,
                            long long sA, long long sB, long long sC,
                            int bDiv, int wtN, int wtTotal) {
  int wt = blockIdx.x * 8 + (threadIdx.x >> 5);
  if (wt >= wtTotal) return;                 // wave-uniform; EXEC stays all-ones
  int lane = threadIdx.x & 31;
  int tN = (wt % wtN) * (16 * WN_);
  int tM = (wt / wtN) * (16 * WM_);
  int batch = blockIdx.y;
  A += (long long)batch * sA;
  W += (long long)(batch / bDiv) * sB;
  C += (long long)batch * sC;

  // A frag (16x32 bf16): lanes0-15 row=lane, elems0-7 = K..K+7, elems8-15 = K+16..23;
  //                      lanes16-31 shifted +8 in K.
  const unsigned short* ap[WM_];
#pragma unroll
  for (int i = 0; i < WM_; ++i) {
    int r = tM + i * 16 + (lane & 15);
    if (r > M - 1) r = M - 1;                // clamp (never triggers: M mult of 32)
    ap[i] = A + (size_t)r * lda + ((lane >> 4) << 3);
  }
  // B frag (32x16 bf16): lanes0-15 col=lane, elems0-7 = K..K+7, elems8-15 = K+16..23; lanes16-31 shifted +8 in K.
  const unsigned short* wp[WN_];
#pragma unroll
  for (int j = 0; j < WN_; ++j) {
    int c = tN + j * 16 + (lane & 15);
    if (c > Wrows - 1) c = Wrows - 1;        // clamp into padded extent
    wp[j] = W + (size_t)c * ldb + ((lane >> 4) << 3);
  }

  const v8f vz = {0.f, 0.f, 0.f, 0.f, 0.f, 0.f, 0.f, 0.f};
  v8f acc[WM_][WN_];
#pragma unroll
  for (int i = 0; i < WM_; ++i)
#pragma unroll
    for (int j = 0; j < WN_; ++j) acc[i][j] = vz;

  for (int k0 = 0; k0 < Kp; k0 += 32) {
    FragBf a[WM_], b[WN_];
#pragma unroll
    for (int i = 0; i < WM_; ++i) {
      a[i].h[0] = *(const us8*)(ap[i] + k0);
      a[i].h[1] = *(const us8*)(ap[i] + k0 + 16);
    }
#pragma unroll
    for (int j = 0; j < WN_; ++j) {
      b[j].h[0] = *(const us8*)(wp[j] + k0);
      b[j].h[1] = *(const us8*)(wp[j] + k0 + 16);
    }
    __builtin_prefetch(ap[0] + k0 + 32, 0, 0);   // global_prefetch_b8, one K-step ahead
#pragma unroll
    for (int i = 0; i < WM_; ++i)
#pragma unroll
      for (int j = 0; j < WN_; ++j)
        acc[i][j] = __builtin_amdgcn_wmma_f32_16x16x32_bf16(false, a[i].v, false, b[j].v,
                                                            (short)0, acc[i][j], false, false);
  }

  // C layout per 16x16 tile: lanes0-15 N=lane, VGPRv -> M=v ; lanes16-31 -> M=v+8
#pragma unroll
  for (int i = 0; i < WM_; ++i) {
#pragma unroll
    for (int j = 0; j < WN_; ++j) {
      int ncol = tN + j * 16 + (lane & 15);
      int mOff = tM + i * 16 + ((lane >> 4) << 3);
      float bv = (bias != nullptr && ncol < N) ? bias[ncol] : 0.f;
#pragma unroll
      for (int v = 0; v < 8; ++v) {
        int m = mOff + v;
        if (m < M && ncol < N) C[(size_t)m * ldc + ncol] = acc[i][j][v] + bv;
      }
    }
  }
}

// ---------------- GRU recurrence (batch rows independent; 16 rows / block) --------
// xp: (T, Nr, 450) row-major rows t*Nr+r (includes bih). out[t,row,colOff+i] = m*h_new.
__global__ void k_gru(const float* __restrict__ xp, const float* __restrict__ Whh,
                      const float* __restrict__ bhh, const float* __restrict__ mask,
                      float* __restrict__ out,
                      int T, int Nr, int ldc, int colOff, int reverse) {
  __shared__ float h[16][H_ + 2];
  __shared__ float hg[16][G3_];
  int tid = threadIdx.x;
  int rowBase = blockIdx.x * 16;
  for (int i = tid; i < 16 * H_; i += blockDim.x) h[i / H_][i % H_] = 0.f;
  __syncthreads();
  for (int st = 0; st < T; ++st) {
    int t = reverse ? (T - 1 - st) : st;
    for (int idx = tid; idx < 16 * G3_; idx += blockDim.x) {
      int r = idx / G3_, g = idx % G3_;
      const float* w = Whh + (size_t)g * H_;
      float acc = bhh[g];
      for (int k2 = 0; k2 < H_; ++k2) acc += h[r][k2] * w[k2];
      hg[r][g] = acc;
    }
    __syncthreads();
    for (int idx = tid; idx < 16 * H_; idx += blockDim.x) {
      int r = idx / H_, ii = idx % H_;
      int row = rowBase + r;
      const float* xg = xp + ((size_t)t * Nr + row) * G3_;
      float rg = sigm(xg[ii] + hg[r][ii]);
      float z  = sigm(xg[H_ + ii] + hg[r][H_ + ii]);
      float nn = tanhf(xg[2 * H_ + ii] + rg * hg[r][2 * H_ + ii]);
      float ho = h[r][ii];
      float hnew = (1.f - z) * nn + z * ho;
      float mk = mask[(size_t)row * T + t];
      h[r][ii] = mk * hnew + (1.f - mk) * ho;
      out[((size_t)t * Nr + row) * ldc + colOff + ii] = mk * hnew;
    }
    __syncthreads();
  }
}

// ---------------- ce (T,B,300) f32 -> bf16 (B,TC,EMBP) t-major -------------------
__global__ void k_ce_to_bf16(const float* __restrict__ ce, unsigned short* __restrict__ o) {
  long long i = (long long)blockIdx.x * blockDim.x + threadIdx.x;
  long long total = (long long)B_ * TC_ * EMBP;
  if (i >= total) return;
  int c = (int)(i % EMBP);
  int t = (int)((i / EMBP) % TC_);
  int b = (int)(i / ((long long)EMBP * TC_));
  float v = (c < D_) ? ce[((size_t)t * B_ + b) * D_ + c] : 0.f;
  o[i] = f2bf(v);
}
// ce -> bf16 transposed (B, DP, TC) d-major (for the "aware" GEMM B operand)
__global__ void k_ce_transpose(const float* __restrict__ ce, unsigned short* __restrict__ o) {
  long long i = (long long)blockIdx.x * blockDim.x + threadIdx.x;
  long long total = (long long)B_ * DP * TC_;
  if (i >= total) return;
  int t = (int)(i % TC_);
  int d = (int)((i / TC_) % DP);
  int b = (int)(i / ((long long)TC_ * DP));
  float v = (d < D_) ? ce[((size_t)t * B_ + b) * D_ + d] : 0.f;
  o[i] = f2bf(v);
}

// ---------------- q_len ----------------
__global__ void k_qlen(const float* __restrict__ q_mask, float* __restrict__ q_len) {
  int b = threadIdx.x;
  if (b < B_) {
    float s = 0.f;
    for (int t = 0; t < TQ_; ++t) s += q_mask[b * TQ_ + t];
    q_len[b] = s;
  }
}

// ---------------- candidate span gather -> qa f32 (L,N,300), qa bf16 (N,L,EMBP), vm --
__global__ void k_qa_gather(const int* __restrict__ cand, const float* __restrict__ ce,
                            const float* __restrict__ qe, const float* __restrict__ q_len,
                            float* __restrict__ qa, unsigned short* __restrict__ qa_bf,
                            float* __restrict__ vm) {
  long long i = (long long)blockIdx.x * blockDim.x + threadIdx.x;
  long long total = (long long)N_ * L_ * EMBP;
  if (i >= total) return;
  int c = (int)(i % EMBP);
  int l = (int)((i / EMBP) % L_);
  int n = (int)(i / ((long long)EMBP * L_));
  int b = n / KK_, k = n % KK_;
  int s = cand[(b * KK_ + k) * 2 + 0];
  int e = cand[(b * KK_ + k) * 2 + 1];
  int alen = e - s;
  int ql = (int)q_len[b];
  float valid = (l < alen + ql) ? 1.f : 0.f;
  float v = 0.f;
  if (c < D_) {
    if (l < alen) {
      int ci = s + l; ci = ci < 0 ? 0 : (ci > TC_ - 1 ? TC_ - 1 : ci);
      v = ce[((size_t)ci * B_ + b) * D_ + c];
    } else {
      int qi = l - alen; qi = qi < 0 ? 0 : (qi > TQ_ - 1 ? TQ_ - 1 : qi);
      v = qe[((size_t)qi * B_ + b) * D_ + c];
    }
    v *= valid;
    qa[((size_t)l * N_ + n) * D_ + c] = v;
  }
  qa_bf[((size_t)n * L_ + l) * EMBP + c] = f2bf(v);
  if (c == 0) vm[n * L_ + l] = valid;
}

// ---------------- masked softmax over T=400 -> alpha bf16 ----------------
__global__ void k_softmax_alpha(const float* __restrict__ scores, const float* __restrict__ cmask,
                                unsigned short* __restrict__ alpha) {
  int lane = threadIdx.x & 31;
  int row = blockIdx.x * 8 + (threadIdx.x >> 5);   // row = n*L + l
  if (row >= N_ * L_) return;
  int b = (row / L_) / KK_;
  const float* s = scores + (size_t)row * TC_;
  const float* cm = cmask + (size_t)b * TC_;
  float mx = -3.0e38f;
  for (int t = lane; t < TC_; t += 32) {
    float v = cm[t] > 0.f ? s[t] : -1e30f;
    mx = fmaxf(mx, v);
  }
  for (int o = 16; o >= 1; o >>= 1) mx = fmaxf(mx, __shfl_xor(mx, o, 32));
  float sum = 0.f;
  for (int t = lane; t < TC_; t += 32) {
    float v = cm[t] > 0.f ? s[t] : -1e30f;
    sum += __expf(v - mx);
  }
  for (int o = 16; o >= 1; o >>= 1) sum += __shfl_xor(sum, o, 32);
  float inv = 1.f / sum;
  unsigned short* a = alpha + (size_t)row * TC_;
  for (int t = lane; t < TC_; t += 32) {
    float v = cm[t] > 0.f ? s[t] : -1e30f;
    a[t] = f2bf(__expf(v - mx) * inv);
  }
}

// ---------------- fused = [qa, aware, qa*aware, qa-aware] bf16 (L*N, 1200) ---------
__global__ void k_fused_build(const float* __restrict__ qa, const float* __restrict__ aware,
                              unsigned short* __restrict__ fused) {
  long long i = (long long)blockIdx.x * blockDim.x + threadIdx.x;
  long long total = (long long)L_ * N_ * D_;
  if (i >= total) return;
  int d = (int)(i % D_);
  long long row = i / D_;                 // row = l*N + n
  int n = (int)(row % N_), l = (int)(row / N_);
  float q = qa[row * D_ + d];
  float a = aware[((size_t)n * L_ + l) * D_ + d];
  unsigned short* f = fused + row * FD_;
  f[d]           = f2bf(q);
  f[D_ + d]      = f2bf(a);
  f[2 * D_ + d]  = f2bf(q * a);
  f[3 * D_ + d]  = f2bf(q - a);
}

// ---------------- m = tanh(fusion out) -> bf16 (L*N, EMBP) zero-padded ------------
__global__ void k_tanh_m(const float* __restrict__ fo, unsigned short* __restrict__ m) {
  long long i = (long long)blockIdx.x * blockDim.x + threadIdx.x;
  long long total = (long long)L_ * N_ * EMBP;
  if (i >= total) return;
  int c = (int)(i % EMBP);
  long long row = i / EMBP;
  float v = (c < D_) ? tanhf(fo[row * D_ + c]) : 0.f;
  m[i] = f2bf(v);
}

// ---------------- masked max-pool over L, dot with outw -> logits (N) -------------
__global__ void k_maxpool_logits(const float* __restrict__ meas, const float* __restrict__ outw,
                                 const float* __restrict__ outb, float* __restrict__ logits) {
  int n = blockIdx.x;
  __shared__ float red[256];
  float acc = 0.f;
  for (int d = threadIdx.x; d < D_; d += blockDim.x) {
    float mx = -3.0e38f;
    for (int l = 0; l < L_; ++l)
      mx = fmaxf(mx, meas[((size_t)l * N_ + n) * D_ + d]);
    acc += mx * outw[d];
  }
  red[threadIdx.x] = acc;
  __syncthreads();
  for (int s = 128; s > 0; s >>= 1) {
    if (threadIdx.x < s) red[threadIdx.x] += red[threadIdx.x + s];
    __syncthreads();
  }
  if (threadIdx.x == 0) logits[n] = red[0] + outb[0];
}

// ---------------- softmax over K=10 per batch ----------------
__global__ void k_softmax_b(const float* __restrict__ logits, float* __restrict__ out) {
  int b = threadIdx.x;
  if (b >= B_) return;
  float mx = -3.0e38f;
  for (int k = 0; k < KK_; ++k) mx = fmaxf(mx, logits[b * KK_ + k]);
  float e[KK_], s = 0.f;
  for (int k = 0; k < KK_; ++k) { e[k] = __expf(logits[b * KK_ + k] - mx); s += e[k]; }
  float inv = 1.f / s;
  for (int k = 0; k < KK_; ++k) out[b * KK_ + k] = e[k] * inv;
}

// ======================================================================
extern "C" void kernel_launch(void* const* d_in, const int* in_sizes, int n_in,
                              void* d_out, int out_size, void* d_ws, size_t ws_size,
                              hipStream_t stream) {
  (void)in_sizes; (void)n_in; (void)out_size; (void)ws_size;
  const int*   context = (const int*)d_in[0];
  const int*   question = (const int*)d_in[1];
  const int*   cand = (const int*)d_in[2];
  const float* emb = (const float*)d_in[3];
  const float* eWihf = (const float*)d_in[4];
  const float* eWhhf = (const float*)d_in[5];
  const float* ebihf = (const float*)d_in[6];
  const float* ebhhf = (const float*)d_in[7];
  const float* eWihb = (const float*)d_in[8];
  const float* eWhhb = (const float*)d_in[9];
  const float* ebihb = (const float*)d_in[10];
  const float* ebhhb = (const float*)d_in[11];
  const float* mWihf = (const float*)d_in[12];
  const float* mWhhf = (const float*)d_in[13];
  const float* mbihf = (const float*)d_in[14];
  const float* mbhhf = (const float*)d_in[15];
  const float* mWihb = (const float*)d_in[16];
  const float* mWhhb = (const float*)d_in[17];
  const float* mbihb = (const float*)d_in[18];
  const float* mbhhb = (const float*)d_in[19];
  const float* fusW = (const float*)d_in[20];
  const float* fusb = (const float*)d_in[21];
  const float* outw = (const float*)d_in[22];
  const float* outb = (const float*)d_in[23];
  float* out = (float*)d_out;

  // ---- workspace carve-up ----
  size_t off = 0;
  auto alloc = [&](size_t bytes) -> char* {
    char* p = (char*)d_ws + off;
    off += (bytes + 255) & ~(size_t)255;
    return p;
  };
  unsigned short* ctxX   = (unsigned short*)alloc((size_t)TC_ * B_ * EMBP * 2);
  unsigned short* qX     = (unsigned short*)alloc((size_t)TQ_ * B_ * EMBP * 2);
  unsigned short* eWfP   = (unsigned short*)alloc((size_t)G3P * EMBP * 2);
  unsigned short* eWbP   = (unsigned short*)alloc((size_t)G3P * EMBP * 2);
  unsigned short* mWfP   = (unsigned short*)alloc((size_t)G3P * EMBP * 2);
  unsigned short* mWbP   = (unsigned short*)alloc((size_t)G3P * EMBP * 2);
  unsigned short* fusWP  = (unsigned short*)alloc((size_t)DP * FD_ * 2);
  float* xp_cf = (float*)alloc((size_t)TC_ * B_ * G3_ * 4);
  float* xp_cb = (float*)alloc((size_t)TC_ * B_ * G3_ * 4);
  float* xp_qf = (float*)alloc((size_t)TQ_ * B_ * G3_ * 4);
  float* xp_qb = (float*)alloc((size_t)TQ_ * B_ * G3_ * 4);
  float* ce    = (float*)alloc((size_t)TC_ * B_ * D_ * 4);
  float* qe    = (float*)alloc((size_t)TQ_ * B_ * D_ * 4);
  float* cmask = (float*)alloc((size_t)B_ * TC_ * 4);
  float* qmask = (float*)alloc((size_t)B_ * TQ_ * 4);
  float* qlen  = (float*)alloc((size_t)B_ * 4);
  float* qa    = (float*)alloc((size_t)L_ * N_ * D_ * 4);
  unsigned short* qaB  = (unsigned short*)alloc((size_t)N_ * L_ * EMBP * 2);
  float* vm    = (float*)alloc((size_t)N_ * L_ * 4);
  unsigned short* ceB  = (unsigned short*)alloc((size_t)B_ * TC_ * EMBP * 2);
  unsigned short* ceT  = (unsigned short*)alloc((size_t)B_ * DP * TC_ * 2);
  float* scores = (float*)alloc((size_t)N_ * L_ * TC_ * 4);
  unsigned short* alphaB = (unsigned short*)alloc((size_t)N_ * L_ * TC_ * 2);
  float* aware  = (float*)alloc((size_t)N_ * L_ * D_ * 4);
  unsigned short* fused = (unsigned short*)alloc((size_t)L_ * N_ * FD_ * 2);
  float* fo     = (float*)alloc((size_t)L_ * N_ * D_ * 4);
  unsigned short* mB = (unsigned short*)alloc((size_t)L_ * N_ * EMBP * 2);
  float* xp_mf = (float*)alloc((size_t)L_ * N_ * G3_ * 4);
  float* xp_mb = (float*)alloc((size_t)L_ * N_ * G3_ * 4);
  float* meas  = (float*)alloc((size_t)L_ * N_ * D_ * 4);
  float* logits = (float*)alloc((size_t)N_ * 4);

  auto cdiv = [](long long a, long long b) { return (int)((a + b - 1) / b); };
  auto gemm = [&](const unsigned short* A, const unsigned short* W, float* C,
                  const float* bias, int M, int N, int Wrows, int Kp,
                  int lda, int ldb, int ldc,
                  long long sA, long long sB, long long sC, int bDiv, int batch) {
    int wtN = (N + 16 * WN_ - 1) / (16 * WN_);
    int wtM = (M + 16 * WM_ - 1) / (16 * WM_);
    int wtTotal = wtM * wtN;
    dim3 g((unsigned)cdiv(wtTotal, 8), (unsigned)batch);
    k_gemm_bf16<<<g, 256, 0, stream>>>(A, W, C, bias, M, N, Wrows, Kp, lda, ldb, ldc,
                                       sA, sB, sC, bDiv, wtN, wtTotal);
  };

  // 1) pad/convert weights to bf16
  k_convert_pad<<<cdiv((long long)G3P * EMBP, 256), 256, 0, stream>>>(eWihf, eWfP, G3_, EMB_, G3P, EMBP);
  k_convert_pad<<<cdiv((long long)G3P * EMBP, 256), 256, 0, stream>>>(eWihb, eWbP, G3_, EMB_, G3P, EMBP);
  k_convert_pad<<<cdiv((long long)G3P * EMBP, 256), 256, 0, stream>>>(mWihf, mWfP, G3_, D_, G3P, EMBP);
  k_convert_pad<<<cdiv((long long)G3P * EMBP, 256), 256, 0, stream>>>(mWihb, mWbP, G3_, D_, G3P, EMBP);
  k_convert_pad<<<cdiv((long long)DP * FD_, 256), 256, 0, stream>>>(fusW, fusWP, D_, FD_, DP, FD_);

  // 2) embedding gather (bf16) + masks
  k_gather_embed<<<cdiv((long long)TC_ * B_ * EMBP, 256), 256, 0, stream>>>(context, emb, ctxX, cmask, TC_, B_);
  k_gather_embed<<<cdiv((long long)TQ_ * B_ * EMBP, 256), 256, 0, stream>>>(question, emb, qX, qmask, TQ_, B_);
  k_qlen<<<1, 32, 0, stream>>>(qmask, qlen);

  // 3) encoder input projections (WMMA): xp = X @ Wih^T + bih
  gemm(ctxX, eWfP, xp_cf, ebihf, TC_ * B_, G3_, G3P, EMBP, EMBP, EMBP, G3_, 0, 0, 0, 1, 1);
  gemm(ctxX, eWbP, xp_cb, ebihb, TC_ * B_, G3_, G3P, EMBP, EMBP, EMBP, G3_, 0, 0, 0, 1, 1);
  gemm(qX,   eWfP, xp_qf, ebihf, TQ_ * B_, G3_, G3P, EMBP, EMBP, EMBP, G3_, 0, 0, 0, 1, 1);
  gemm(qX,   eWbP, xp_qb, ebihb, TQ_ * B_, G3_, G3P, EMBP, EMBP, EMBP, G3_, 0, 0, 0, 1, 1);

  // 4) encoder GRU recurrences (batch-parallel blocks of 16 rows)
  k_gru<<<B_ / 16, 256, 0, stream>>>(xp_cf, eWhhf, ebhhf, cmask, ce, TC_, B_, D_, 0, 0);
  k_gru<<<B_ / 16, 256, 0, stream>>>(xp_cb, eWhhb, ebhhb, cmask, ce, TC_, B_, D_, H_, 1);
  k_gru<<<B_ / 16, 256, 0, stream>>>(xp_qf, eWhhf, ebhhf, qmask, qe, TQ_, B_, D_, 0, 0);
  k_gru<<<B_ / 16, 256, 0, stream>>>(xp_qb, eWhhb, ebhhb, qmask, qe, TQ_, B_, D_, H_, 1);

  // 5) pack ce to bf16 (t-major and d-major)
  k_ce_to_bf16<<<cdiv((long long)B_ * TC_ * EMBP, 256), 256, 0, stream>>>(ce, ceB);
  k_ce_transpose<<<cdiv((long long)B_ * DP * TC_, 256), 256, 0, stream>>>(ce, ceT);

  // 6) candidate span gather
  k_qa_gather<<<cdiv((long long)N_ * L_ * EMBP, 256), 256, 0, stream>>>(cand, ce, qe, qlen, qa, qaB, vm);

  // 7) attention scores (batched WMMA, one batch per n, ce shared per b=n/10)
  gemm(qaB, ceB, scores, nullptr, L_, TC_, TC_, EMBP, EMBP, EMBP, TC_,
       (long long)L_ * EMBP, (long long)TC_ * EMBP, (long long)L_ * TC_, KK_, N_);

  // 8) masked softmax -> alpha bf16
  k_softmax_alpha<<<(N_ * L_) / 8, 256, 0, stream>>>(scores, cmask, alphaB);

  // 9) aware = alpha @ ce (batched WMMA)
  gemm(alphaB, ceT, aware, nullptr, L_, D_, DP, TC_, TC_, TC_, D_,
       (long long)L_ * TC_, (long long)DP * TC_, (long long)L_ * D_, KK_, N_);

  // 10) fusion features + fusion GEMM + tanh
  k_fused_build<<<cdiv((long long)L_ * N_ * D_, 256), 256, 0, stream>>>(qa, aware, fused);
  gemm(fused, fusWP, fo, fusb, L_ * N_, D_, DP, FD_, FD_, FD_, D_, 0, 0, 0, 1, 1);
  k_tanh_m<<<cdiv((long long)L_ * N_ * EMBP, 256), 256, 0, stream>>>(fo, mB);

  // 11) match GRU input projections (WMMA) + recurrences
  gemm(mB, mWfP, xp_mf, mbihf, L_ * N_, G3_, G3P, EMBP, EMBP, EMBP, G3_, 0, 0, 0, 1, 1);
  gemm(mB, mWbP, xp_mb, mbihb, L_ * N_, G3_, G3P, EMBP, EMBP, EMBP, G3_, 0, 0, 0, 1, 1);
  k_gru<<<N_ / 16, 256, 0, stream>>>(xp_mf, mWhhf, mbhhf, vm, meas, L_, N_, D_, 0, 0);
  k_gru<<<N_ / 16, 256, 0, stream>>>(xp_mb, mWhhb, mbhhb, vm, meas, L_, N_, D_, H_, 1);

  // 12) max-pool + output head + softmax over K
  k_maxpool_logits<<<N_, 256, 0, stream>>>(meas, outw, outb, logits);
  k_softmax_b<<<1, 32, 0, stream>>>(logits, out);
}
